// _MSAPairWeightedAveraging_34995393528037
// MI455X (gfx1250) — matrix operand order repaired and checked by reading, not compiled, not run on hardware
//
#include <hip/hip_runtime.h>
#include <hip/hip_bf16.h>
#include <math.h>

typedef __attribute__((ext_vector_type(16))) _Float16 v16h;
typedef __attribute__((ext_vector_type(8)))  _Float16 v8h;
typedef __attribute__((ext_vector_type(8)))  float    v8f;

#define S_  512
#define N_  512
#define CM_ 64
#define CZ_ 128
#define H_  8
#define D_  8
#define EPS_ 1e-5f

__device__ __forceinline__ float wave_sum(float v) {
#pragma unroll
  for (int m = 16; m >= 1; m >>= 1) v += __shfl_xor(v, m, 32);
  return v;
}
__device__ __forceinline__ float wave_max(float v) {
#pragma unroll
  for (int m = 16; m >= 1; m >>= 1) v = fmaxf(v, __shfl_xor(v, m, 32));
  return v;
}

__device__ __forceinline__ v16h cat8(v8h lo, v8h hi) {
  return __builtin_shufflevector(lo, hi, 0, 1, 2, 3, 4, 5, 6, 7,
                                 8, 9, 10, 11, 12, 13, 14, 15);
}

// ---------------------------------------------------------------------------
// k1: m = LN(m_si); v = m@W_v^T (store transposed f16: vT[h][n*8+d][j]);
//     g = sigmoid(m@W_g^T) (store f16 row-major [row, cm]).
// One wave per (j,n) row of 64 channels; 8 waves / block.
// ---------------------------------------------------------------------------
__global__ __launch_bounds__(256) void k1_ln_vg(
    const float* __restrict__ m_si, const float* __restrict__ lnw,
    const float* __restrict__ lnb, const float* __restrict__ Wv,
    const float* __restrict__ Wg, _Float16* __restrict__ vT,
    _Float16* __restrict__ g_ws) {
  __shared__ float sWv[CM_ * CM_];
  __shared__ float sWg[CM_ * CM_];
  __shared__ float sm[8][CM_];
  int tid = threadIdx.x;
  for (int t = tid; t < CM_ * CM_; t += 256) { sWv[t] = Wv[t]; sWg[t] = Wg[t]; }
  int wid = tid >> 5, lane = tid & 31;
  int row = blockIdx.x * 8 + wid;          // 0..S*N-1
  int c0  = lane * 2;
  float2 mv = *(const float2*)(m_si + (size_t)row * CM_ + c0);
  float s  = wave_sum(mv.x + mv.y);
  float sq = wave_sum(mv.x * mv.x + mv.y * mv.y);
  float mu = s * (1.0f / CM_);
  float rstd = rsqrtf(sq * (1.0f / CM_) - mu * mu + EPS_);
  float n0 = (mv.x - mu) * rstd * lnw[c0]     + lnb[c0];
  float n1 = (mv.y - mu) * rstd * lnw[c0 + 1] + lnb[c0 + 1];
  sm[wid][c0] = n0; sm[wid][c0 + 1] = n1;
  __syncthreads();
  float a0 = 0.f, a1 = 0.f, b0 = 0.f, b1 = 0.f;
#pragma unroll 8
  for (int k = 0; k < CM_; ++k) {
    float mk = sm[wid][k];
    a0 = fmaf(mk, sWv[c0 * CM_ + k], a0);
    a1 = fmaf(mk, sWv[(c0 + 1) * CM_ + k], a1);
    b0 = fmaf(mk, sWg[c0 * CM_ + k], b0);
    b1 = fmaf(mk, sWg[(c0 + 1) * CM_ + k], b1);
  }
  float g0 = 1.0f / (1.0f + expf(-b0));
  float g1 = 1.0f / (1.0f + expf(-b1));
  g_ws[(size_t)row * CM_ + c0]     = (_Float16)g0;
  g_ws[(size_t)row * CM_ + c0 + 1] = (_Float16)g1;
  int j = row >> 9;          // row / N_
  int n = row & (N_ - 1);
  vT[((size_t)(c0 >> 3) * (N_ * D_) + n * D_ + (c0 & 7)) * S_ + j] = (_Float16)a0;
  int c1 = c0 + 1;
  vT[((size_t)(c1 >> 3) * (N_ * D_) + n * D_ + (c1 & 7)) * S_ + j] = (_Float16)a1;
}

// ---------------------------------------------------------------------------
// k2: b = LN(z)@W_b^T; w = softmax_j(b); store wT[h][i][j] f16.
// One block per i; waves sweep j; then wave w soft-maxes head w.
// ---------------------------------------------------------------------------
__global__ __launch_bounds__(256) void k2_bias_softmax(
    const float* __restrict__ z, const float* __restrict__ lnw,
    const float* __restrict__ lnb, const float* __restrict__ Wb,
    _Float16* __restrict__ wT) {
  __shared__ float sWb[H_ * CZ_];
  __shared__ float slnw[CZ_], slnb[CZ_];
  __shared__ float slog[S_ * H_];           // 16 KB
  int tid = threadIdx.x;
  for (int t = tid; t < H_ * CZ_; t += 256) sWb[t] = Wb[t];
  for (int t = tid; t < CZ_; t += 256) { slnw[t] = lnw[t]; slnb[t] = lnb[t]; }
  __syncthreads();
  int wid = tid >> 5, lane = tid & 31;
  int i = blockIdx.x;
  const float* zrow = z + (size_t)i * S_ * CZ_;
  for (int j = wid; j < S_; j += 8) {
    float4 zv = *(const float4*)(zrow + (size_t)j * CZ_ + lane * 4);
    float s  = wave_sum(zv.x + zv.y + zv.z + zv.w);
    float sq = wave_sum(zv.x * zv.x + zv.y * zv.y + zv.z * zv.z + zv.w * zv.w);
    float mu = s * (1.0f / CZ_);
    float rstd = rsqrtf(sq * (1.0f / CZ_) - mu * mu + EPS_);
    int c = lane * 4;
    float nz0 = (zv.x - mu) * rstd * slnw[c]     + slnb[c];
    float nz1 = (zv.y - mu) * rstd * slnw[c + 1] + slnb[c + 1];
    float nz2 = (zv.z - mu) * rstd * slnw[c + 2] + slnb[c + 2];
    float nz3 = (zv.w - mu) * rstd * slnw[c + 3] + slnb[c + 3];
#pragma unroll
    for (int h = 0; h < H_; ++h) {
      float p = nz0 * sWb[h * CZ_ + c]     + nz1 * sWb[h * CZ_ + c + 1]
              + nz2 * sWb[h * CZ_ + c + 2] + nz3 * sWb[h * CZ_ + c + 3];
      p = wave_sum(p);
      if (lane == 0) slog[j * H_ + h] = p;
    }
  }
  __syncthreads();
  int h = wid;                               // 8 waves = 8 heads
  float l[16];
  float mx = -1e30f;
#pragma unroll
  for (int t = 0; t < 16; ++t) {
    l[t] = slog[(lane + t * 32) * H_ + h];
    mx = fmaxf(mx, l[t]);
  }
  mx = wave_max(mx);
  float sum = 0.f;
#pragma unroll
  for (int t = 0; t < 16; ++t) { l[t] = expf(l[t] - mx); sum += l[t]; }
  sum = wave_sum(sum);
  float r = 1.0f / sum;
#pragma unroll
  for (int t = 0; t < 16; ++t)
    wT[((size_t)h * S_ + i) * S_ + lane + t * 32] = (_Float16)(l[t] * r);
}

// ---------------------------------------------------------------------------
// k3: o[i,c] (per head) = sum_j wT[h][i][j] * vT[h][c][j] via WMMA f16->f32.
// Register-blocked: each wave owns a 32x64 macro-tile (2 i-tiles x 4 c-tiles,
// 8 accumulators). Per k-step: 2 A-frags + 4 B-frags feed 8 WMMAs
// (1.5 b128 loads per WMMA). Fused go = g*o epilogue, f16 row-major.
// A k order used here (contiguous): lane<16 -> K 0..15; lane>=16 -> K 16..31
// B k order used here (contiguous, same as A): lane<16 -> K 0..15; lane>=16 -> K 16..31 (col=lane&15)
// ---------------------------------------------------------------------------
__global__ __launch_bounds__(256) void k3_pwa_wmma(
    const _Float16* __restrict__ wT, const _Float16* __restrict__ vT,
    const _Float16* __restrict__ g_ws, _Float16* __restrict__ go) {
  int wid = threadIdx.x >> 5, lane = threadIdx.x & 31;
  int gw  = blockIdx.x * 8 + wid;           // 0..8191
  int h   = gw >> 10;                        // 1024 macro-tiles per head
  int rem = gw & 1023;
  int itg = rem >> 6;                        // 16 groups of 2 i-tiles
  int ctg = rem & 63;                        // 64 groups of 4 c-tiles
  int half = (lane < 16) ? 0 : 1;
  int l15  = lane & 15;
  int koffA = half * 16;
  int koffB = half * 16;

  const _Float16* A0 = wT + ((size_t)h * S_ + itg * 32 + l15) * S_;
  const _Float16* A1 = A0 + (size_t)16 * S_;
  const _Float16* B0 = vT + ((size_t)h * (N_ * D_) + ctg * 64 + l15) * S_;
  const _Float16* B1 = B0 + (size_t)16 * S_;
  const _Float16* B2 = B0 + (size_t)32 * S_;
  const _Float16* B3 = B0 + (size_t)48 * S_;

  v8f acc[2][4];
#pragma unroll
  for (int a = 0; a < 2; ++a)
#pragma unroll
    for (int b = 0; b < 4; ++b) acc[a][b] = (v8f){};

#pragma unroll 4
  for (int kt = 0; kt < S_; kt += 32) {
    v16h a0 = *(const v16h*)(A0 + kt + koffA);
    v16h a1 = *(const v16h*)(A1 + kt + koffA);
    v16h b0 = *(const v16h*)(B0 + kt + koffB);
    v16h b1 = *(const v16h*)(B1 + kt + koffB);
    v16h b2 = *(const v16h*)(B2 + kt + koffB);
    v16h b3 = *(const v16h*)(B3 + kt + koffB);
    acc[0][0] = __builtin_amdgcn_wmma_f32_16x16x32_f16(false, a0, false, b0,
                                                       (short)0, acc[0][0], false, false);
    acc[0][1] = __builtin_amdgcn_wmma_f32_16x16x32_f16(false, a0, false, b1,
                                                       (short)0, acc[0][1], false, false);
    acc[0][2] = __builtin_amdgcn_wmma_f32_16x16x32_f16(false, a0, false, b2,
                                                       (short)0, acc[0][2], false, false);
    acc[0][3] = __builtin_amdgcn_wmma_f32_16x16x32_f16(false, a0, false, b3,
                                                       (short)0, acc[0][3], false, false);
    acc[1][0] = __builtin_amdgcn_wmma_f32_16x16x32_f16(false, a1, false, b0,
                                                       (short)0, acc[1][0], false, false);
    acc[1][1] = __builtin_amdgcn_wmma_f32_16x16x32_f16(false, a1, false, b1,
                                                       (short)0, acc[1][1], false, false);
    acc[1][2] = __builtin_amdgcn_wmma_f32_16x16x32_f16(false, a1, false, b2,
                                                       (short)0, acc[1][2], false, false);
    acc[1][3] = __builtin_amdgcn_wmma_f32_16x16x32_f16(false, a1, false, b3,
                                                       (short)0, acc[1][3], false, false);
  }

  // C/D layout: VGPR v holds (M = v + 8*half, N = lane&15)
#pragma unroll
  for (int u = 0; u < 4; ++u) {
    int c = ctg * 64 + u * 16 + l15;
    int n = c >> 3, d = c & 7;
    size_t colbase = (size_t)n * CM_ + h * D_ + d;
#pragma unroll
    for (int a = 0; a < 2; ++a) {
#pragma unroll
      for (int v = 0; v < 8; ++v) {
        int irow = itg * 32 + a * 16 + v + half * 8;
        size_t idx = (size_t)irow * ((size_t)N_ * CM_) + colbase;
        float gv = (float)g_ws[idx];
        go[idx] = (_Float16)(gv * acc[a][u][v]);
      }
    }
  }
}

// ---------------------------------------------------------------------------
// k4: out = go @ W_out^T via WMMA (K=64 -> 2 steps); W_out cvt f32->f16 inline.
// out[r, m] = sum_k go[r,k] * W_out[m,k]; B[k,m] = W_out[m,k] contiguous in k.
// ---------------------------------------------------------------------------
__global__ __launch_bounds__(256) void k4_out_proj(
    const _Float16* __restrict__ go, const float* __restrict__ Wout,
    float* __restrict__ out) {
  int wid = threadIdx.x >> 5, lane = threadIdx.x & 31;
  int gw  = blockIdx.x * 8 + wid;           // 0..65535
  int rt  = gw >> 2;                         // 16384 row tiles
  int mt  = gw & 3;                          // 4 col tiles
  int half = (lane < 16) ? 0 : 1;
  int l15  = lane & 15;
  const _Float16* Arow = go + (size_t)(rt * 16 + l15) * CM_;
  const float*    Brow = Wout + (size_t)(mt * 16 + l15) * CM_;
  int koffA = half * 16;
  int koffB = half * 16;
  v8f acc = {};
#pragma unroll
  for (int kt = 0; kt < CM_; kt += 32) {
    v16h a = *(const v16h*)(Arow + kt + koffA);
    float4 f0 = *(const float4*)(Brow + kt + koffB);
    float4 f1 = *(const float4*)(Brow + kt + koffB + 4);
    float4 f2 = *(const float4*)(Brow + kt + koffB + 8);
    float4 f3 = *(const float4*)(Brow + kt + koffB + 12);
    v16h b;
    b[0]  = (_Float16)f0.x; b[1]  = (_Float16)f0.y;
    b[2]  = (_Float16)f0.z; b[3]  = (_Float16)f0.w;
    b[4]  = (_Float16)f1.x; b[5]  = (_Float16)f1.y;
    b[6]  = (_Float16)f1.z; b[7]  = (_Float16)f1.w;
    b[8]  = (_Float16)f2.x; b[9]  = (_Float16)f2.y;
    b[10] = (_Float16)f2.z; b[11] = (_Float16)f2.w;
    b[12] = (_Float16)f3.x; b[13] = (_Float16)f3.y;
    b[14] = (_Float16)f3.z; b[15] = (_Float16)f3.w;
    acc = __builtin_amdgcn_wmma_f32_16x16x32_f16(false, a, false, b,
                                                 (short)0, acc, false, false);
  }
#pragma unroll
  for (int v = 0; v < 8; ++v) {
    int r = rt * 16 + v + half * 8;
    out[(size_t)r * CM_ + mt * 16 + l15] = acc[v];
  }
}

// ---------------------------------------------------------------------------
extern "C" void kernel_launch(void* const* d_in, const int* in_sizes, int n_in,
                              void* d_out, int out_size, void* d_ws, size_t ws_size,
                              hipStream_t stream) {
  (void)in_sizes; (void)n_in; (void)out_size; (void)ws_size;
  const float* m_si   = (const float*)d_in[0];
  const float* z_ij   = (const float*)d_in[1];
  const float* ln_m_w = (const float*)d_in[2];
  const float* ln_m_b = (const float*)d_in[3];
  const float* ln_z_w = (const float*)d_in[4];
  const float* ln_z_b = (const float*)d_in[5];
  const float* W_v    = (const float*)d_in[6];
  const float* W_b    = (const float*)d_in[7];
  const float* W_g    = (const float*)d_in[8];
  const float* W_out  = (const float*)d_in[9];

  char* ws = (char*)d_ws;
  _Float16* vT   = (_Float16*)(ws);                        // 8*4096*512*2 = 32 MiB
  _Float16* wT   = (_Float16*)(ws + 33554432);             // 8*512*512*2  =  4 MiB
  _Float16* g_ws = (_Float16*)(ws + 37748736);             // 512*512*64*2 = 32 MiB
  _Float16* go   = (_Float16*)(ws + 71303168);             // 512*512*64*2 = 32 MiB
  float* out = (float*)d_out;

  k1_ln_vg<<<32768, 256, 0, stream>>>(m_si, ln_m_w, ln_m_b, W_v, W_g, vT, g_ws);
  k2_bias_softmax<<<512, 256, 0, stream>>>(z_ij, ln_z_w, ln_z_b, W_b, wT);
  k3_pwa_wmma<<<1024, 256, 0, stream>>>(wT, vT, g_ws, go);
  k4_out_proj<<<8192, 256, 0, stream>>>(go, W_out, out);
}
